// TranslationRegressor_83150566851358
// MI455X (gfx1250) — compile-verified, matrix operand order checked by reading, not run on hardware
//
#include <hip/hip_runtime.h>
#include <stdint.h>

// ---------------------------------------------------------------------------
// CDNA5 (gfx1250) wave32 WMMA types
// ---------------------------------------------------------------------------
typedef __attribute__((ext_vector_type(16))) __bf16 v16bf;
typedef __attribute__((ext_vector_type(8)))  float  v8f;

#define NEG_SLOPE 0.01f

__device__ __forceinline__ float leakyf(float x) { return x >= 0.f ? x : NEG_SLOPE * x; }

// round-to-nearest-even f32 -> bf16 (bit pattern as ushort)
__device__ __forceinline__ unsigned short f2bf(float x) {
  union { float f; unsigned u; } c; c.f = x;
  unsigned u = c.u;
  unsigned r = (u + 0x7FFFu + ((u >> 16) & 1u)) >> 16;
  return (unsigned short)r;
}

// ---------------------------------------------------------------------------
// Async global->LDS staging via inline asm (GLOBAL_LOAD_ASYNC_TO_LDS_B128).
// The clang builtin is declared with CUDA-LangAS pointer params that HIP
// source cannot produce, so we go straight to the ISA (bridge-doc guidance).
//   - %0: 32-bit VGPR = LDS byte address (low 32 bits of generic shared ptr)
//   - %1: 64-bit VGPR pair = global VA (GV addressing mode, saddr = off)
// Tracked by ASYNCcnt; completed with s_wait_asynccnt before the barrier.
// ---------------------------------------------------------------------------
__device__ __forceinline__ void async_copy_b128(const unsigned short* g, const unsigned short* l) {
  unsigned lds_off = (unsigned)(uintptr_t)(const void*)l;
  unsigned long long ga = (unsigned long long)(uintptr_t)(const void*)g;
  asm volatile("global_load_async_to_lds_b128 %0, %1, off"
               :
               : "v"(lds_off), "v"(ga)
               : "memory");
}
__device__ __forceinline__ void wait_async0() {
#if defined(__has_builtin)
#if __has_builtin(__builtin_amdgcn_s_wait_asynccnt)
  __builtin_amdgcn_s_wait_asynccnt(0);
#else
  asm volatile("s_wait_asynccnt 0x0" ::: "memory");
#endif
#else
  asm volatile("s_wait_asynccnt 0x0" ::: "memory");
#endif
}

// ---------------------------------------------------------------------------
// Convert kernels (one-time per launch; trivial traffic vs. GEMMs)
// ---------------------------------------------------------------------------

// Weight transpose+convert: W is (Ktot x N) f32 row-major; out[n*K + k] = bf16(W[(row0+k)*N + n])
__global__ void kconv_wT(const float* __restrict__ W, int N, int row0, int K,
                         unsigned short* __restrict__ out, int total) {
  int i = blockIdx.x * blockDim.x + threadIdx.x;
  if (i >= total) return;
  int n = i / K, k = i - n * K;
  out[i] = f2bf(W[(size_t)(row0 + k) * N + n]);
}

__global__ void kconv_copy_bf16(const float* __restrict__ src,
                                unsigned short* __restrict__ dst, int total) {
  int i = blockIdx.x * blockDim.x + threadIdx.x;
  if (i >= total) return;
  dst[i] = f2bf(src[i]);
}

// anchor rows: dst[b*2144 + k] = bf16(features[(b*32 + 0)*2144 + k])
__global__ void kconv_anchor(const float* __restrict__ src,
                             unsigned short* __restrict__ dst, int total) {
  int i = blockIdx.x * blockDim.x + threadIdx.x;
  if (i >= total) return;
  int b = i / 2144, k = i - b * 2144;
  dst[i] = f2bf(src[(size_t)b * 32 * 2144 + k]);
}

// ---------------------------------------------------------------------------
// bf16 WMMA GEMM:  C(MxN) = A(MxK) * Bt(NxK)^T  + fused epilogue
//   block tile 128x128, 256 threads (8 waves: 4 in M x 2 in N)
//   wave tile 32x64 = 2x4 WMMA 16x16 tiles, K-step 32
//   double-buffered LDS, async global->LDS staging, one barrier per k-step
// ---------------------------------------------------------------------------
union Frag { v16bf v; uint4 u[2]; };

__global__ __launch_bounds__(256)
void kgemm_bf16(const unsigned short* __restrict__ A,    // M x K bf16
                const unsigned short* __restrict__ Bt,   // N x K bf16 (pre-transposed)
                int M, int N, int K,
                const float* __restrict__ bias,          // N, or null
                const float* __restrict__ anchor,        // (M/32) x N f32, or null
                const float* __restrict__ residual,      // M x N f32, or null
                float* __restrict__ outF32, int leakyF32,
                unsigned short* __restrict__ outBF16, int leakyBF16) {
  __shared__ unsigned short As[2][128][40];  // +8 pad keeps 16B align, spreads banks
  __shared__ unsigned short Bs[2][128][40];

  const int tid  = threadIdx.x;
  const int lane = tid & 31;
  const int wave = tid >> 5;
  const int wm   = wave & 3;        // wave position in M (0..3)
  const int wn   = wave >> 2;       // wave position in N (0..1)
  const int bx   = blockIdx.x * 128;
  const int by   = blockIdx.y * 128;
  const int ldr  = tid >> 1;        // staging row 0..127
  const int ldc  = (tid & 1) * 16;  // staging col 0 or 16
  const int hi   = lane >> 4;       // lane half (ISA 16-bit A/B layout)
  const int ln   = lane & 15;

  const unsigned short* gA = A  + (size_t)(by + ldr) * K + ldc;
  const unsigned short* gB = Bt + (size_t)(bx + ldr) * K + ldc;

  v8f acc[2][4] = {};

  const int NT = K >> 5;  // K / 32 k-tiles
  int bufi = 0;

  // ---- prologue: stage tile 0 into buffer 0 ----
  async_copy_b128(gA,     &As[0][ldr][ldc]);
  async_copy_b128(gA + 8, &As[0][ldr][ldc + 8]);
  async_copy_b128(gB,     &Bs[0][ldr][ldc]);
  async_copy_b128(gB + 8, &Bs[0][ldr][ldc + 8]);
  wait_async0();
  __syncthreads();

  for (int t = 0; t < NT; ++t) {
    const int havenext = (t + 1 < NT);
    const int nxt = bufi ^ 1;
    const int koff = (t + 1) << 5;

    // ---- stage next tile into the other buffer (overlaps with WMMA below) ----
    if (havenext) {
      async_copy_b128(gA + koff,     &As[nxt][ldr][ldc]);
      async_copy_b128(gA + koff + 8, &As[nxt][ldr][ldc + 8]);
      async_copy_b128(gB + koff,     &Bs[nxt][ldr][ldc]);
      async_copy_b128(gB + koff + 8, &Bs[nxt][ldr][ldc + 8]);
    }

    // ---- fragments per ISA 16-bit layout: elem e -> K = 8*hi + (e&7) + (e>=8 ? 16 : 0) ----
    Frag fa[2], fb[4];
#pragma unroll
    for (int mi = 0; mi < 2; ++mi) {
      int m = wm * 32 + mi * 16 + ln;
      fa[mi].u[0] = *(const uint4*)&As[bufi][m][8 * hi];
      fa[mi].u[1] = *(const uint4*)&As[bufi][m][16 + 8 * hi];
    }
#pragma unroll
    for (int ni = 0; ni < 4; ++ni) {
      int n = wn * 64 + ni * 16 + ln;
      fb[ni].u[0] = *(const uint4*)&Bs[bufi][n][8 * hi];
      fb[ni].u[1] = *(const uint4*)&Bs[bufi][n][16 + 8 * hi];
    }
#pragma unroll
    for (int mi = 0; mi < 2; ++mi)
#pragma unroll
      for (int ni = 0; ni < 4; ++ni)
        acc[mi][ni] = __builtin_amdgcn_wmma_f32_16x16x32_bf16(
            false, fa[mi].v, false, fb[ni].v, (short)0, acc[mi][ni], false, false);

    wait_async0();          // my async stage into nxt complete before signaling
    __syncthreads();        // everyone done reading bufi + staging nxt visible
    bufi = nxt;
  }

  // ---- fused epilogue: bias + anchor-broadcast + residual + leaky, dual-dtype store ----
#pragma unroll
  for (int mi = 0; mi < 2; ++mi) {
#pragma unroll
    for (int ni = 0; ni < 4; ++ni) {
      const int colg = bx + wn * 64 + ni * 16 + ln;
      const float bval = bias ? bias[colg] : 0.f;
#pragma unroll
      for (int rr = 0; rr < 8; ++rr) {
        const int rowg = by + wm * 32 + mi * 16 + hi * 8 + rr;
        float v = acc[mi][ni][rr] + bval;
        if (anchor)   v += anchor[(size_t)(rowg >> 5) * N + colg];  // per-batch anchor term
        if (residual) v += residual[(size_t)rowg * N + colg];
        const size_t o = (size_t)rowg * N + colg;
        if (outF32)  outF32[o]  = leakyF32 ? leakyf(v) : v;
        if (outBF16) outBF16[o] = f2bf(leakyBF16 ? leakyf(v) : v);
      }
    }
  }
}

// ---------------------------------------------------------------------------
// Tiny 512 -> 3 head (WMMA would waste 125/128 of the N tile)
// ---------------------------------------------------------------------------
__global__ __launch_bounds__(256)
void khead(const float* __restrict__ a5, const float* __restrict__ Wd,
           const float* __restrict__ bd, float* __restrict__ out, int M) {
  __shared__ float w[512 * 3];
  for (int i = threadIdx.x; i < 512 * 3; i += blockDim.x) w[i] = Wd[i];
  __syncthreads();
  int row = blockIdx.x * blockDim.x + threadIdx.x;
  if (row >= M) return;
  float s0 = bd[0], s1 = bd[1], s2 = bd[2] + 1.0f;  // + [0,0,1] bias
  const float* x = a5 + (size_t)row * 512;
  for (int k = 0; k < 512; ++k) {
    float xv = x[k];
    s0 += xv * w[k * 3 + 0];
    s1 += xv * w[k * 3 + 1];
    s2 += xv * w[k * 3 + 2];
  }
  out[row * 3 + 0] = s0;
  out[row * 3 + 1] = s1;
  out[row * 3 + 2] = s2;
}

// ---------------------------------------------------------------------------
// Launch
// ---------------------------------------------------------------------------
extern "C" void kernel_launch(void* const* d_in, const int* in_sizes, int n_in,
                              void* d_out, int out_size, void* d_ws, size_t ws_size,
                              hipStream_t stream) {
  (void)in_sizes; (void)n_in; (void)out_size; (void)ws_size;

  const float* features = (const float*)d_in[0];
  const float* We  = (const float*)d_in[1];
  const float* be  = (const float*)d_in[2];
  const float* W1a = (const float*)d_in[3];
  const float* b1a = (const float*)d_in[4];
  const float* W1b = (const float*)d_in[5];
  const float* b1b = (const float*)d_in[6];
  const float* W2a = (const float*)d_in[7];
  const float* b2a = (const float*)d_in[8];
  const float* W2b = (const float*)d_in[9];
  const float* b2b = (const float*)d_in[10];
  const float* Wd  = (const float*)d_in[11];
  const float* bd  = (const float*)d_in[12];
  float* out = (float*)d_out;

  const int MROWS = 16384;  // B*T
  const int D  = 2144;
  const int H1 = 1024;
  const int H2 = 512;
  const int NB = 512;       // batches (anchor rows)

  // workspace carve-out (aliased buffers noted)
  char* wsp = (char*)d_ws;
  size_t off = 0;
  auto take = [&](size_t bytes) -> void* {
    void* p = wsp + off;
    off += (bytes + 255) & ~(size_t)255;
    return p;
  };
  unsigned short* WeTopT = (unsigned short*)take((size_t)H1 * D * 2);
  unsigned short* WeBotT = (unsigned short*)take((size_t)H1 * D * 2);
  unsigned short* W1aT   = (unsigned short*)take((size_t)H1 * H1 * 2);
  unsigned short* W1bT   = (unsigned short*)take((size_t)H2 * H1 * 2);
  unsigned short* W2aT   = (unsigned short*)take((size_t)H2 * H2 * 2);
  unsigned short* W2bT   = (unsigned short*)take((size_t)H2 * H2 * 2);
  unsigned short* Xanc   = (unsigned short*)take((size_t)NB * D * 2);
  float*          ancE   = (float*)take((size_t)NB * H1 * 4);
  unsigned short* Xtok   = (unsigned short*)take((size_t)MROWS * D * 2);   // reused as a2
  unsigned short* a1     = (unsigned short*)take((size_t)MROWS * H1 * 2);  // reused as a5f
  float*          hbuf   = (float*)take((size_t)MROWS * H2 * 4);
  unsigned short* a3     = (unsigned short*)take((size_t)MROWS * H2 * 2);
  unsigned short* a4     = (unsigned short*)take((size_t)MROWS * H2 * 2);
  unsigned short* a2  = Xtok;        // Xtok dead after L1
  float*          a5f = (float*)a1;  // a1 dead after L2; 16384*1024*2B == 16384*512*4B

  const int TH = 256;
  // --- weight transpose+convert ---
  { int t = H1 * D;  kconv_wT<<<(t + TH - 1) / TH, TH, 0, stream>>>(We,  H1, 0, D,  WeTopT, t); }
  { int t = H1 * D;  kconv_wT<<<(t + TH - 1) / TH, TH, 0, stream>>>(We,  H1, D, D,  WeBotT, t); }
  { int t = H1 * H1; kconv_wT<<<(t + TH - 1) / TH, TH, 0, stream>>>(W1a, H1, 0, H1, W1aT,  t); }
  { int t = H2 * H1; kconv_wT<<<(t + TH - 1) / TH, TH, 0, stream>>>(W1b, H2, 0, H1, W1bT,  t); }
  { int t = H2 * H2; kconv_wT<<<(t + TH - 1) / TH, TH, 0, stream>>>(W2a, H2, 0, H2, W2aT,  t); }
  { int t = H2 * H2; kconv_wT<<<(t + TH - 1) / TH, TH, 0, stream>>>(W2b, H2, 0, H2, W2bT,  t); }
  // --- activation convert ---
  { int t = MROWS * D; kconv_copy_bf16<<<(t + TH - 1) / TH, TH, 0, stream>>>(features, Xtok, t); }
  { int t = NB * D;    kconv_anchor<<<(t + TH - 1) / TH, TH, 0, stream>>>(features, Xanc, t); }

  dim3 blk(256);
  // anchor term: ancE = Xanc @ We[2144:]   (512 x 1024, f32, no bias/act)
  kgemm_bf16<<<dim3(H1 / 128, NB / 128), blk, 0, stream>>>(
      Xanc, WeBotT, NB, H1, D, nullptr, nullptr, nullptr, ancE, 0, nullptr, 0);
  // L1: a1 = leaky(Xtok @ We[:2144] + ancE[b] + be)
  kgemm_bf16<<<dim3(H1 / 128, MROWS / 128), blk, 0, stream>>>(
      Xtok, WeTopT, MROWS, H1, D, be, ancE, nullptr, nullptr, 0, a1, 1);
  // L2: a2 = leaky(a1 @ W1a + b1a)
  kgemm_bf16<<<dim3(H1 / 128, MROWS / 128), blk, 0, stream>>>(
      a1, W1aT, MROWS, H1, H1, b1a, nullptr, nullptr, nullptr, 0, a2, 1);
  // L3: hbuf = a2 @ W1b + b1b (f32, block1 out);  a3 = leaky(hbuf) (bf16)
  kgemm_bf16<<<dim3(H2 / 128, MROWS / 128), blk, 0, stream>>>(
      a2, W1bT, MROWS, H2, H1, b1b, nullptr, nullptr, hbuf, 0, a3, 1);
  // L4: a4 = leaky(a3 @ W2a + b2a)
  kgemm_bf16<<<dim3(H2 / 128, MROWS / 128), blk, 0, stream>>>(
      a3, W2aT, MROWS, H2, H2, b2a, nullptr, nullptr, nullptr, 0, a4, 1);
  // L5: a5f = leaky(a4 @ W2b + b2b + hbuf)   (residual add, f32)
  kgemm_bf16<<<dim3(H2 / 128, MROWS / 128), blk, 0, stream>>>(
      a4, W2bT, MROWS, H2, H2, b2b, nullptr, hbuf, a5f, 1, nullptr, 0);
  // head: out = a5f @ Wd + bd + [0,0,1]
  khead<<<(MROWS + TH - 1) / TH, TH, 0, stream>>>(a5f, Wd, bd, out, MROWS);
}
